// LogicGatedSNN_54434415510063
// MI455X (gfx1250) — matrix operand order repaired and checked by reading, not compiled, not run on hardware
//
#include <hip/hip_runtime.h>

#define IN_F   8192
#define OUT_F  8192
#define KSPLIT 1024          // per-wave K range: IN_F / 8 waves
#define THRESH 50.0f

typedef __attribute__((ext_vector_type(16))) _Float16 v16h;
typedef __attribute__((ext_vector_type(8)))  _Float16 v8h;
typedef __attribute__((ext_vector_type(4)))  _Float16 v4h;
typedef __attribute__((ext_vector_type(8)))  float    v8f;

// One workgroup (256 threads = 8 wave32) computes 16 output rows.
// Each wave owns a 1024-wide K slice and accumulates with
// v_wmma_f32_16x16x32_f16; partial D fragments are LDS-reduced.
__global__ __launch_bounds__(256)
void snn_gemv_wmma(const float* __restrict__ x,      // [IN_F] spikes {0,1}
                   const float* __restrict__ S,      // [OUT_F, IN_F] states
                   const float* __restrict__ mem,    // [OUT_F]
                   const float* __restrict__ thr,    // [OUT_F]
                   const float* __restrict__ noise,  // [OUT_F]
                   float* __restrict__ out)          // [OUT_F] spikes
{
    __shared__ _Float16 xh[IN_F];     // spike vector as exact f16 {0,1}
    __shared__ float    red[8 * 16];  // per-wave partial currents

    const int tid     = threadIdx.x;
    const int lane    = tid & 31;
    const int wave    = tid >> 5;
    const int rowBase = blockIdx.x * 16;

    // ---- Stage spike input into LDS as f16 (exact for 0.0/1.0) ----
    #pragma unroll
    for (int i = 0; i < IN_F / (256 * 4); ++i) {   // 8 iters, float4 per thread
        const int idx = (i * 256 + tid) * 4;
        const float4 v = *reinterpret_cast<const float4*>(x + idx);
        v4h h;
        h[0] = (_Float16)v.x; h[1] = (_Float16)v.y;
        h[2] = (_Float16)v.z; h[3] = (_Float16)v.w;
        *reinterpret_cast<v4h*>(&xh[idx]) = h;     // 8B LDS store
    }
    __syncthreads();

    // ---- Per-wave GEMV slice via WMMA ----
    // B is K x N = 32 x 16, N = output row. B column n is row (rowBase+n) of S,
    // contiguous in K: lanes 0-15 hold K=0..15 of col=lane, lanes 16-31 hold
    // K=16..31 of col=lane-16  -> 64B contiguous load per lane.
    // A is M x K = 16 x 32 with every row equal to x (so D row 0 is the GEMV):
// lanes 0-15 hold K=0..15; lanes 16-31 hold
// K=16..31 (same contiguous order as B).
    const int    kBase = wave * KSPLIT;
    const float* wRow  = S + (size_t)(rowBase + (lane & 15)) * IN_F;
    const int    aSel  = (lane >= 16) ? 16 : 0;
    const int    bSel  = (lane >= 16) ? 16 : 0;

    v8f c = {};
    #pragma unroll 2
    for (int kk = 0; kk < KSPLIT; kk += 32) {
        const int k0 = kBase + kk;

        // A fragment from LDS (two ds_load_b128)
        const v8h lo = *reinterpret_cast<const v8h*>(&xh[k0 + aSel]);
        const v8h hi = *reinterpret_cast<const v8h*>(&xh[k0 + aSel + 8]);
        v16h a;
        #pragma unroll
        for (int i = 0; i < 8; ++i) { a[i] = lo[i]; a[8 + i] = hi[i]; }

        // B fragment: 16 consecutive fp32 states, binarized to f16 {0,1}
        const float4* wp = reinterpret_cast<const float4*>(wRow + k0 + bSel);
        // keep the stream ahead of the loads
        __builtin_prefetch(wRow + k0 + bSel + 512, 0, 0);
        v16h b;
        #pragma unroll
        for (int v = 0; v < 4; ++v) {
            const float4 q = wp[v];
            b[4 * v + 0] = (q.x > THRESH) ? (_Float16)1.0f : (_Float16)0.0f;
            b[4 * v + 1] = (q.y > THRESH) ? (_Float16)1.0f : (_Float16)0.0f;
            b[4 * v + 2] = (q.z > THRESH) ? (_Float16)1.0f : (_Float16)0.0f;
            b[4 * v + 3] = (q.w > THRESH) ? (_Float16)1.0f : (_Float16)0.0f;
        }

        c = __builtin_amdgcn_wmma_f32_16x16x32_f16(
                /*neg_a=*/false, a, /*neg_b=*/false, b,
                /*c_mod=*/(short)0, c, /*reuse_a=*/false, /*reuse_b=*/false);
    }

    // ---- Reduce the 8 per-wave partials and apply the neuron update ----
    // D layout: VGPR0 of lanes 0..15 holds M=0, N=lane  (the GEMV result)
    if (lane < 16) red[wave * 16 + lane] = c[0];
    __syncthreads();

    if (tid < 16) {
        float cur = 0.0f;
        #pragma unroll
        for (int w = 0; w < 8; ++w) cur += red[w * 16 + tid];
        const int   o   = rowBase + tid;
        const float pot = mem[o] + cur + noise[o];
        out[o] = (pot >= thr[o]) ? 1.0f : 0.0f;
    }
}

extern "C" void kernel_launch(void* const* d_in, const int* in_sizes, int n_in,
                              void* d_out, int out_size, void* d_ws, size_t ws_size,
                              hipStream_t stream) {
    const float* x     = (const float*)d_in[0];   // spike_input      [1, IN_F]
    const float* S     = (const float*)d_in[1];   // synapse_states   [OUT_F, IN_F]
    const float* mem   = (const float*)d_in[2];   // membrane_potential [OUT_F]
    const float* thr   = (const float*)d_in[3];   // adaptive_threshold [OUT_F]
    const float* noise = (const float*)d_in[4];   // noise              [OUT_F]
    float* out = (float*)d_out;                   // spikes             [OUT_F]

    dim3 grid(OUT_F / 16);   // 512 workgroups x 16 rows
    dim3 block(256);         // 8 wave32 per workgroup
    snn_gemv_wmma<<<grid, block, 0, stream>>>(x, S, mem, thr, noise, out);
}
